// ViTStack_37538014167631
// MI455X (gfx1250) — matrix operand order repaired and checked by reading, not compiled, not run on hardware
//
#include <hip/hip_runtime.h>

typedef _Float16 half_t;
typedef __attribute__((ext_vector_type(16))) _Float16 v16h;
typedef __attribute__((ext_vector_type(8)))  _Float16 v8h_t;
typedef __attribute__((ext_vector_type(4)))  _Float16 v4h_t;
typedef __attribute__((ext_vector_type(8)))  float    v8f;
typedef __attribute__((ext_vector_type(4)))  float    v4f;
typedef __attribute__((ext_vector_type(4)))  unsigned int v4u;
typedef __attribute__((ext_vector_type(8)))  int      v8i;
typedef __attribute__((ext_vector_type(4)))  int      v4i;

// ---------------- problem constants ----------------
constexpr int Bc   = 8;
constexpr int Cc   = 256;
constexpr int Nc   = 4096;      // 64*64 tokens per image
constexpr int BNc  = Bc * Nc;   // 32768 token rows
constexpr int Hc   = 8;
constexpr int HDc  = 32;
constexpr int Lc   = 8;
constexpr int FFc  = 768;
constexpr int NCHc = 64;        // chunks per (b,h)
constexpr int CHc  = 64;        // chunk length
constexpr float LOG1E4 = 9.210340371976184f;

// ---------------- TDM availability ------------------------------------
#if __has_builtin(__builtin_amdgcn_tensor_load_to_lds)
#define USE_TDM 1
#if __has_include(<hip/amd_detail/amd_gfx1250_TDM.h>)
#define TDM_6ARG 1
#endif
#endif

// ---------------- WMMA fragment helpers (contiguous k order, k = 16*(lane/16) + i) ----
static __device__ __forceinline__ v8f wmma16(v16h a, v16h b, v8f c) {
  return __builtin_amdgcn_wmma_f32_16x16x32_f16(false, a, false, b, (short)0, c,
                                                false, false);
}

// A fragment 16x32 (MxK) from row-major matrix, leading-dim ld (halves).
// lanes 0-15: row=lane, K {0..15}; lanes 16-31: row=lane-16, K {16..31}
static __device__ __forceinline__ v16h load_a16(const half_t* base, int ld,
                                                int row0, int k0, int lane) {
  int hs = lane >> 4;
  const half_t* p = base + (row0 + (lane & 15)) * ld + k0;
  v16h a;
#pragma unroll
  for (int i = 0; i < 8; ++i) a[i] = p[hs * 16 + i];
#pragma unroll
  for (int i = 0; i < 8; ++i) a[8 + i] = p[hs * 16 + 8 + i];
  return a;
}

// B fragment 32x16 (KxN) from row-major K-major matrix, leading-dim ld.
static __device__ __forceinline__ v16h load_b16(const half_t* base, int ld,
                                                int k0, int col0, int lane) {
  int hs = lane >> 4;
  int c = col0 + (lane & 15);
  v16h b;
#pragma unroll
  for (int i = 0; i < 16; ++i) b[i] = base[(k0 + hs * 16 + i) * ld + c];
  return b;
}

// B fragment where source is stored N-major (we need B = srcT): src[n][k]
static __device__ __forceinline__ v16h load_bt16(const half_t* base, int ld,
                                                 int k0, int col0, int lane) {
  int hs = lane >> 4;
  const half_t* p = base + (col0 + (lane & 15)) * ld + k0 + hs * 16;
  v16h b;
#pragma unroll
  for (int i = 0; i < 16; ++i) b[i] = p[i];
  return b;
}

// ---------------- generic WMMA GEMM -----------------------------------
constexpr int BMt = 128, BNt = 64, BKt = 32;
constexpr int LDA_S = BKt + 8;   // 40 halves = 80B row (TDM pad produces this)
constexpr int LDB_S = BNt + 8;   // 72

#ifdef USE_TDM
// Issue one TDM 2-D tile load: 128 rows x 32 halves from row-major A(K stride)
// into LDS with 16B padding after every 64B row chunk (=> 40-half row stride).
// D# layout per CDNA5 ISA.
static __device__ __forceinline__ void tdm_load_a_tile(const half_t* gsrc,
                                                       unsigned lds_off,
                                                       int K, int M) {
  unsigned long long ga = (unsigned long long)(size_t)gsrc;
  v4u g0 = {1u,                                   // count=1, user mode
            lds_off,                              // lds_addr
            (unsigned)ga,                         // global_addr[31:0]
            (unsigned)((ga >> 32) & 0x1FFFFFFu) | (2u << 30)};  // [56:32]|type=2
  v8i g1 = {(int)((1u << 16)      // data_size = 2 bytes
                  | (1u << 20)    // pad_enable
                  | (3u << 22)    // pad_interval: 16 DWORDs (64B)
                  | (3u << 25)),  // pad_amount:   4 DWORDs (16B)
            (int)((unsigned)(K & 0xFFFF) << 16),                 // tensor_dim0 lo
            (int)(((unsigned)K >> 16) | ((unsigned)(M & 0xFFFF) << 16)),
            (int)(((unsigned)M >> 16) | (32u << 16)),            // tile_dim0=32
            128,                                                 // tile_dim1=128
            K,                                                   // dim0_stride lo
            0, 0};
  v4i gz = {0, 0, 0, 0};
#ifdef TDM_6ARG
  v8i gz8 = {0, 0, 0, 0, 0, 0, 0, 0};
  __builtin_amdgcn_tensor_load_to_lds(g0, g1, gz, gz, gz8, 0);
#else
  __builtin_amdgcn_tensor_load_to_lds(g0, g1, gz, gz, 0);
#endif
}
#endif

// out(MxN) = A(MxK,f16) * B(KxN) + bias ; B source f32 (opt. transposed NxK).
//   EPI 0: out_h = f16(acc+bias)
//   EPI 1: out_f32 += acc+bias              (residual add)
//   EPI 2: out_h = f16(aux_h * relu(acc+bias))
//   EPI 3: out_f32 = acc+bias
template <int EPI, bool BT>
__global__ __launch_bounds__(256) void gemm_kernel(
    const half_t* __restrict__ A, const float* __restrict__ Bsrc,
    const float* __restrict__ bias, int M, int N, int K,
    float* __restrict__ out_f32, half_t* __restrict__ out_h,
    const half_t* __restrict__ aux_h) {
  __shared__ __align__(16) half_t As[2][BMt][LDA_S];
  __shared__ __align__(16) half_t Bs[BKt][LDB_S];

  const int tid  = threadIdx.x;
  const int lane = tid & 31;
  const int wid  = tid >> 5;          // 8 waves
  const int wr   = wid >> 1;          // 0..3  (row block of 32)
  const int wc   = wid & 1;           // 0..1  (col block of 32)
  const int m0   = blockIdx.y * BMt;
  const int n0   = blockIdx.x * BNt;

  v8f acc00 = {0,0,0,0,0,0,0,0}, acc01 = {0,0,0,0,0,0,0,0};
  v8f acc10 = {0,0,0,0,0,0,0,0}, acc11 = {0,0,0,0,0,0,0,0};

#ifdef USE_TDM
  if (wid == 0)   // prologue: DMA tile 0 into buffer 0
    tdm_load_a_tile(A + (size_t)m0 * K, (unsigned)(size_t)&As[0][0][0], K, M);
#endif

  const int nsteps = K / BKt;
  for (int it = 0; it < nsteps; ++it) {
    const int k0  = it * BKt;
    const int cur = it & 1;

    // stage B tile (f32 -> f16); vectorized when K-major
    if (!BT) {
#pragma unroll
      for (int ch = tid; ch < (BKt * BNt) / 4; ch += 256) {
        int r = ch >> 4, c = (ch & 15) * 4;
        v4f v = *(const v4f*)&Bsrc[(size_t)(k0 + r) * N + n0 + c];
        v4h_t h4;
#pragma unroll
        for (int i = 0; i < 4; ++i) h4[i] = (half_t)v[i];
        *(v4h_t*)&Bs[r][c] = h4;
      }
    } else {
#pragma unroll
      for (int idx = tid; idx < BKt * BNt; idx += 256) {
        int r = idx / BNt, c = idx - r * BNt;
        Bs[r][c] = (half_t)Bsrc[(size_t)(n0 + c) * K + (k0 + r)];
      }
    }

#ifdef USE_TDM
    if (wid == 0) {
      if (it + 1 < nsteps) {   // overlap next tile's DMA with this compute
        tdm_load_a_tile(A + (size_t)m0 * K + (k0 + BKt),
                        (unsigned)(size_t)&As[cur ^ 1][0][0], K, M);
        __builtin_amdgcn_s_wait_tensorcnt(1);   // in-order: current tile done
      } else {
        __builtin_amdgcn_s_wait_tensorcnt(0);
      }
    }
#else
    // synchronous fallback: 16B global loads -> 16B LDS stores
#pragma unroll
    for (int ch = tid; ch < (BMt * BKt) / 8; ch += 256) {
      int r = ch >> 2, c = (ch & 3) * 8;
      *(v8h_t*)&As[cur][r][c] = *(const v8h_t*)&A[(size_t)(m0 + r) * K + k0 + c];
    }
    if (k0 + BKt < K)
      __builtin_prefetch(&A[(size_t)(m0 + (tid >> 1)) * K + k0 + BKt], 0, 0);
#endif
    __syncthreads();

    v16h af0 = load_a16(&As[cur][0][0], LDA_S, wr * 32,      0, lane);
    v16h af1 = load_a16(&As[cur][0][0], LDA_S, wr * 32 + 16, 0, lane);
    v16h bf0 = load_b16(&Bs[0][0], LDB_S, 0, wc * 32,      lane);
    v16h bf1 = load_b16(&Bs[0][0], LDB_S, 0, wc * 32 + 16, lane);
    acc00 = wmma16(af0, bf0, acc00);
    acc01 = wmma16(af0, bf1, acc01);
    acc10 = wmma16(af1, bf0, acc10);
    acc11 = wmma16(af1, bf1, acc11);
    __syncthreads();
  }

  const int hs = lane >> 4, cl = lane & 15;
  v8f* accs[2][2] = {{&acc00, &acc01}, {&acc10, &acc11}};
#pragma unroll
  for (int im = 0; im < 2; ++im)
#pragma unroll
    for (int jn = 0; jn < 2; ++jn) {
      int r0 = m0 + wr * 32 + im * 16;
      int col = n0 + wc * 32 + jn * 16 + cl;
      float bc_ = bias[col];
      v8f a = *accs[im][jn];
#pragma unroll
      for (int e = 0; e < 8; ++e) {
        size_t o = (size_t)(r0 + e + hs * 8) * N + col;
        float v = a[e] + bc_;
        if (EPI == 0)      out_h[o]  = (half_t)v;
        else if (EPI == 1) out_f32[o] += v;
        else if (EPI == 2) out_h[o]  = (half_t)((float)aux_h[o] * fmaxf(v, 0.f));
        else               out_f32[o] = v;
      }
    }
}

template <int EPI, bool BT>
static void launch_gemm(const half_t* A, const float* Bsrc, const float* bias,
                        int M, int N, int K, float* outf, half_t* outh,
                        const half_t* aux, hipStream_t st) {
  dim3 grid(N / BNt, M / BMt);
  gemm_kernel<EPI, BT><<<grid, 256, 0, st>>>(A, Bsrc, bias, M, N, K, outf, outh, aux);
}

// ---------------- embeddings ------------------------------------------
__global__ __launch_bounds__(256) void emb_kernel(const float* __restrict__ t,
                                                  half_t* __restrict__ emb) {
  int idx = blockIdx.x * 256 + threadIdx.x;     // BNc*512 total
  int c = idx & 511;
  int token = idx >> 9;
  int b = token >> 12, n = token & 4095;
  int y = n >> 6, xq = n & 63;
  float val;
  if (c < 256) {                                // 2D positional
    int cc = c & 127;
    float pos = (c < 128) ? (float)y : (float)xq;
    int j = cc & 63;
    float f = __expf(-LOG1E4 * (float)j / 64.f);
    float a = pos * f;
    val = (cc < 64) ? __sinf(a) : __cosf(a);
  } else {                                      // time
    int cc = c - 256;
    int j = cc & 127;
    float f = __expf(-LOG1E4 * (float)j / 128.f);
    float a = t[b] * f;
    val = (cc < 128) ? __sinf(a) : __cosf(a);
  }
  emb[idx] = (half_t)val;
}

__global__ __launch_bounds__(256) void encode_apply_kernel(
    const float* __restrict__ x, const float* __restrict__ proj,
    float* __restrict__ h) {
  int idx = blockIdx.x * 256 + threadIdx.x;     // BNc*Cc
  int c = idx & 255;
  int token = idx >> 8;
  int b = token >> 12, n = token & 4095;
  float mul = proj[(size_t)token * 512 + c];
  float bia = proj[(size_t)token * 512 + 256 + c];
  h[idx] = x[((size_t)(b * 256 + c)) * 4096 + n] * mul + bia;
}

// ---------------- layernorm (ddof=1), wave per row --------------------
__global__ __launch_bounds__(256) void ln_kernel(const float* __restrict__ h,
                                                 const float* __restrict__ s,
                                                 const float* __restrict__ bb,
                                                 half_t* __restrict__ u) {
  int row = blockIdx.x * 8 + (threadIdx.x >> 5);
  int lane = threadIdx.x & 31;
  const float* x = h + (size_t)row * Cc;
  float vals[8], sum = 0.f;
#pragma unroll
  for (int i = 0; i < 8; ++i) { vals[i] = x[lane + i * 32]; sum += vals[i]; }
#pragma unroll
  for (int off = 16; off > 0; off >>= 1) sum += __shfl_xor(sum, off, 32);
  float mean = sum * (1.f / Cc);
  float vs = 0.f;
#pragma unroll
  for (int i = 0; i < 8; ++i) { float d = vals[i] - mean; vs += d * d; }
#pragma unroll
  for (int off = 16; off > 0; off >>= 1) vs += __shfl_xor(vs, off, 32);
  float inv = rsqrtf(vs * (1.f / (Cc - 1)) + 1e-6f);
#pragma unroll
  for (int i = 0; i < 8; ++i) {
    int c = lane + i * 32;
    u[(size_t)row * Cc + c] = (half_t)((vals[i] - mean) * inv * s[c] + bb[c]);
  }
}

// ---------------- LSH hashing: codes = argmax([rot,-rot]) -------------
__global__ __launch_bounds__(256) void hash_kernel(const half_t* __restrict__ k_h,
                                                   const float* __restrict__ R,
                                                   int* __restrict__ codes) {
  int bh = blockIdx.x;                 // b*Hc + head
  int b = bh >> 3, hd = bh & 7;
  __shared__ float Rs[HDc][33];
  for (int idx = threadIdx.x; idx < HDc * 32; idx += 256) {
    int d = idx >> 5, r = idx & 31;
    Rs[d][r] = R[(size_t)(hd * HDc + d) * 32 + r];
  }
  __syncthreads();
  for (int n = threadIdx.x; n < Nc; n += 256) {
    const half_t* kp = k_h + ((size_t)(b * Nc + n)) * Cc + hd * HDc;
    float kd[HDc];
#pragma unroll
    for (int d = 0; d < HDc; ++d) kd[d] = (float)kp[d];
    float rot[32];
#pragma unroll
    for (int r = 0; r < 32; ++r) {
      float acc = 0.f;
#pragma unroll
      for (int d = 0; d < HDc; ++d) acc += kd[d] * Rs[d][r];
      rot[r] = acc;
    }
    float best = rot[0]; int bi = 0;
#pragma unroll
    for (int r = 1; r < 32; ++r) if (rot[r] > best) { best = rot[r]; bi = r; }
#pragma unroll
    for (int r = 0; r < 32; ++r) if (-rot[r] > best) { best = -rot[r]; bi = 32 + r; }
    codes[(size_t)bh * Nc + n] = bi;
  }
}

// ---------------- stable counting sort by bucket ----------------------
__global__ __launch_bounds__(64) void sort_kernel(const int* __restrict__ codes,
                                                  int* __restrict__ order) {
  int bh = blockIdx.x;
  __shared__ int hist[64][64];   // [chunk][bucket]
  __shared__ int total[64];
  __shared__ int base[64];
  int t = threadIdx.x;
  const int* cp = codes + (size_t)bh * Nc;
  for (int bkt = 0; bkt < 64; ++bkt) hist[t][bkt] = 0;
  for (int j = 0; j < 64; ++j) hist[t][cp[t * 64 + j]]++;
  __syncthreads();
  {   // thread t owns bucket t: exclusive prefix over chunks
    int run = 0;
    for (int ck = 0; ck < 64; ++ck) { int v = hist[ck][t]; hist[ck][t] = run; run += v; }
    total[t] = run;
  }
  __syncthreads();
  if (t == 0) { int a = 0; for (int bkt = 0; bkt < 64; ++bkt) { base[bkt] = a; a += total[bkt]; } }
  __syncthreads();
  int* op = order + (size_t)bh * Nc;
  for (int j = 0; j < 64; ++j) {     // in-chunk j order => stable
    int n = t * 64 + j;
    int bkt = cp[n];
    int pos = base[bkt] + hist[t][bkt]++;
    op[pos] = n;
  }
}

// ---------------- chunked attention (WMMA) ----------------------------
__global__ __launch_bounds__(128) void attn_kernel(
    const half_t* __restrict__ q_h, const half_t* __restrict__ k_h,
    const half_t* __restrict__ v_h, const int* __restrict__ order,
    half_t* __restrict__ attn_out) {
  int blk = blockIdx.x;
  int ch = blk & (NCHc - 1);
  int bh = blk / NCHc;
  int hd = bh & 7, b = bh >> 3;
  __shared__ int tok[CHc];
  __shared__ __align__(16) half_t qs[CHc][40], ks[CHc][40], vs[CHc][40];
  __shared__ float S[CHc][65];
  __shared__ __align__(16) half_t P[CHc][72];
  int tid = threadIdx.x, lane = tid & 31, w = tid >> 5;

  if (tid < CHc) tok[tid] = order[(size_t)bh * Nc + ch * CHc + tid];
  __syncthreads();
  for (int idx = tid; idx < CHc * HDc; idx += 128) {
    int p = idx >> 5, d = idx & 31;
    size_t base_ = ((size_t)(b * Nc + tok[p])) * Cc + hd * HDc + d;
    qs[p][d] = q_h[base_];
    ks[p][d] = k_h[base_];
    vs[p][d] = v_h[base_];
  }
  __syncthreads();

  // S = Q K^T / sqrt(HD): wave w -> rows [16w,16w+16), 4 col tiles, K=32
  {
    v16h a = load_a16(&qs[0][0], 40, w * 16, 0, lane);
#pragma unroll
    for (int j = 0; j < 4; ++j) {
      v16h bf = load_bt16(&ks[0][0], 40, 0, j * 16, lane);
      v8f acc = {0,0,0,0,0,0,0,0};
      acc = wmma16(a, bf, acc);
      int hs = lane >> 4, c = lane & 15;
#pragma unroll
      for (int e = 0; e < 8; ++e)
        S[w * 16 + e + hs * 8][j * 16 + c] = acc[e] * 0.17677669529663687f;
    }
  }
  __syncthreads();

  if (tid < CHc) {   // softmax over 64 keys
    float mx = -1e30f;
    for (int j = 0; j < CHc; ++j) mx = fmaxf(mx, S[tid][j]);
    float sm = 0.f;
    for (int j = 0; j < CHc; ++j) { float e = __expf(S[tid][j] - mx); S[tid][j] = e; sm += e; }
    float inv = 1.f / sm;
    for (int j = 0; j < CHc; ++j) P[tid][j] = (half_t)(S[tid][j] * inv);
  }
  __syncthreads();

  // O = P V : wave w -> rows [16w,16w+16), 2 col tiles of 16, K=64 (2 WMMAs)
#pragma unroll
  for (int jt = 0; jt < 2; ++jt) {
    v8f acc = {0,0,0,0,0,0,0,0};
#pragma unroll
    for (int kt = 0; kt < 2; ++kt) {
      v16h a = load_a16(&P[0][0], 72, w * 16, kt * 32, lane);
      v16h bf = load_b16(&vs[0][0], 40, kt * 32, jt * 16, lane);
      acc = wmma16(a, bf, acc);
    }
    int hs = lane >> 4, c = lane & 15;
#pragma unroll
    for (int e = 0; e < 8; ++e) {
      int row = w * 16 + e + hs * 8;
      attn_out[((size_t)(b * Nc + tok[row])) * Cc + hd * HDc + jt * 16 + c] =
          (half_t)acc[e];
    }
  }
}

// ---------------- final NCHW transpose --------------------------------
__global__ __launch_bounds__(256) void out_kernel(const float* __restrict__ h,
                                                  float* __restrict__ out) {
  int idx = blockIdx.x * 256 + threadIdx.x;   // BNc*Cc
  int c = idx & 255;
  int token = idx >> 8;
  int b = token >> 12, n = token & 4095;
  out[((size_t)(b * 256 + c)) * 4096 + n] = h[(size_t)token * Cc + c];
}

// ---------------- host orchestration ----------------------------------
extern "C" void kernel_launch(void* const* d_in, const int* in_sizes, int n_in,
                              void* d_out, int out_size, void* d_ws,
                              size_t ws_size, hipStream_t stream) {
  (void)in_sizes; (void)n_in; (void)out_size; (void)ws_size;
  const float* x     = (const float*)d_in[0];
  const float* t     = (const float*)d_in[1];
  const float* enc_W = (const float*)d_in[3];
  const float* enc_b = (const float*)d_in[4];
  const float* ln1_s = (const float*)d_in[5];
  const float* ln1_b = (const float*)d_in[6];
  const float* Wq = (const float*)d_in[7];   const float* bq = (const float*)d_in[8];
  const float* Wk = (const float*)d_in[9];   const float* bk = (const float*)d_in[10];
  const float* Wv = (const float*)d_in[11];  const float* bv = (const float*)d_in[12];
  const float* Wo = (const float*)d_in[13];  const float* bo = (const float*)d_in[14];
  const float* R  = (const float*)d_in[15];
  const float* ln2_s = (const float*)d_in[16]; const float* ln2_b = (const float*)d_in[17];
  const float* Wa = (const float*)d_in[18];  const float* ba = (const float*)d_in[19];
  const float* Wb = (const float*)d_in[20];  const float* bb = (const float*)d_in[21];
  const float* Wc = (const float*)d_in[22];  const float* bc = (const float*)d_in[23];

  // workspace carve (byte offsets, all 256-aligned)
  char* ws = (char*)d_ws;
  float*  h_f32  = (float*)(ws + 0);                       // 33,554,432 B
  half_t* u_h    = (half_t*)(ws + 33554432ull);            // 16,777,216
  half_t* q_h    = (half_t*)(ws + 50331648ull);            // 16,777,216
  half_t* k_h    = (half_t*)(ws + 67108864ull);            // 16,777,216
  half_t* v_h    = (half_t*)(ws + 83886080ull);            // 16,777,216
  half_t* attn_h = (half_t*)(ws + 100663296ull);           // 16,777,216
  int*    codes  = (int*)   (ws + 117440512ull);           //  1,048,576
  int*    order  = (int*)   (ws + 118489088ull);           //  1,048,576
  char*   big    = ws + 119537664ull;                      // 100,663,296 shared
  half_t* emb_h  = (half_t*)(big);                         // enc: 33,554,432
  float*  proj   = (float*) (big + 33554432ull);           // enc: 67,108,864
  half_t* a_h    = (half_t*)(big);                         // ff:  50,331,648
  half_t* ff1    = (half_t*)(big + 50331648ull);           // ff:  50,331,648

  // ---- encoder: emb -> proj (512x512 GEMM, B transposed) -> modulate x
  emb_kernel<<<(BNc * 512) / 256, 256, 0, stream>>>(t, emb_h);
  launch_gemm<3, true>(emb_h, enc_W, enc_b, BNc, 512, 512, proj, nullptr,
                       nullptr, stream);
  encode_apply_kernel<<<(BNc * Cc) / 256, 256, 0, stream>>>(x, proj, h_f32);

  for (int l = 0; l < Lc; ++l) {
    const float* wq = Wq + (size_t)l * Cc * Cc;  const float* bq_ = bq + l * Cc;
    const float* wk = Wk + (size_t)l * Cc * Cc;  const float* bk_ = bk + l * Cc;
    const float* wv = Wv + (size_t)l * Cc * Cc;  const float* bv_ = bv + l * Cc;
    const float* wo = Wo + (size_t)l * Cc * Cc;  const float* bo_ = bo + l * Cc;
    const float* wa = Wa + (size_t)l * Cc * FFc; const float* ba_ = ba + l * FFc;
    const float* wb = Wb + (size_t)l * Cc * FFc; const float* bb_ = bb + l * FFc;
    const float* wc = Wc + (size_t)l * FFc * Cc; const float* bc_ = bc + l * Cc;
    const float* Rl = R + (size_t)l * Hc * HDc * 32;

    // attention block
    ln_kernel<<<BNc / 8, 256, 0, stream>>>(h_f32, ln1_s + l * Cc, ln1_b + l * Cc, u_h);
    launch_gemm<0, false>(u_h, wq, bq_, BNc, Cc, Cc, nullptr, q_h, nullptr, stream);
    launch_gemm<0, false>(u_h, wk, bk_, BNc, Cc, Cc, nullptr, k_h, nullptr, stream);
    launch_gemm<0, false>(u_h, wv, bv_, BNc, Cc, Cc, nullptr, v_h, nullptr, stream);
    hash_kernel<<<Bc * Hc, 256, 0, stream>>>(k_h, Rl, codes);
    sort_kernel<<<Bc * Hc, 64, 0, stream>>>(codes, order);
    attn_kernel<<<Bc * Hc * NCHc, 128, 0, stream>>>(q_h, k_h, v_h, order, attn_h);
    launch_gemm<1, false>(attn_h, wo, bo_, BNc, Cc, Cc, h_f32, nullptr, nullptr, stream);

    // gated FF block
    ln_kernel<<<BNc / 8, 256, 0, stream>>>(h_f32, ln2_s + l * Cc, ln2_b + l * Cc, u_h);
    launch_gemm<0, false>(u_h, wa, ba_, BNc, FFc, Cc, nullptr, a_h, nullptr, stream);
    launch_gemm<2, false>(u_h, wb, bb_, BNc, FFc, Cc, nullptr, ff1, a_h, stream);
    launch_gemm<1, false>(ff1, wc, bc_, BNc, Cc, FFc, h_f32, nullptr, nullptr, stream);
  }

  out_kernel<<<(BNc * Cc) / 256, 256, 0, stream>>>(h_f32, (float*)d_out);
}
